// IterModel_7988639170679
// MI455X (gfx1250) — matrix operand order repaired and checked by reading, not compiled, not run on hardware
//
#include <hip/hip_runtime.h>
#include <hip/hip_bf16.h>

typedef __attribute__((ext_vector_type(16))) __bf16 v16bf;
typedef __attribute__((ext_vector_type(8)))  __bf16 v8bf;
typedef __attribute__((ext_vector_type(8)))  float  v8f;

#define B_SZ 16
#define L_SZ 64
#define H_SZ 512
#define NLAB 64
#define ITERS 3

__device__ __forceinline__ unsigned short f2bf(float x) {
    unsigned int u = __float_as_uint(x);
    unsigned int r = (u + 0x7FFFu + ((u >> 16) & 1u)) >> 16;   // RNE
    return (unsigned short)r;
}

// ---------------------------------------------------------------------------
// Generic WMMA GEMM: C[M,N] = A[M,K] * W[N,K]^T (+ bias[N]), A/W bf16, C fp32.
// One 16x16 C tile per wave; A/B use contiguous k order (k = 16h + s); C layout per CDNA5 ISA 7.12.2.
// ---------------------------------------------------------------------------
__global__ __launch_bounds__(256) void wmma_gemm_xt(
    const unsigned short* __restrict__ A, const unsigned short* __restrict__ W,
    const float* __restrict__ bias, float* __restrict__ C,
    int M, int N, int K)
{
    int lane = threadIdx.x & 31;
    int wave = threadIdx.x >> 5;
    int tiles_n = N >> 4;
    int total = (M >> 4) * tiles_n;
    int t = blockIdx.x * 8 + wave;
    if (t >= total) return;                 // wave-uniform: EXEC stays all-1s
    int mt = t / tiles_n;
    int nt = t - mt * tiles_n;
    int sel  = lane >> 4;                   // lane half selects K sub-range
    int arow = mt * 16 + (lane & 15);
    int bcol = nt * 16 + (lane & 15);       // B column n == row n of W (contiguous)

    v8f acc = {};
    int nkt = K >> 5;
    const __bf16* Ab = (const __bf16*)A;
    const __bf16* Wb = (const __bf16*)W;
    for (int kt = 0; kt < nkt; ++kt) {
        // A fragment: lanes 0-15 row m hold K 0..15; lanes 16-31 hold K 16..31
        union { v16bf v; v8bf h[2]; } a;
        const __bf16* pa = Ab + arow * K + kt * 32 + sel * 16;
        a.h[0] = *(const v8bf*)pa;
        a.h[1] = *(const v8bf*)(pa + 8);
        // B fragment: lanes 0-15 col n hold K 0..15; lanes 16-31 hold K 16..31
        v16bf b = *(const v16bf*)(Wb + bcol * K + kt * 32 + sel * 16);
        acc = __builtin_amdgcn_wmma_f32_16x16x32_bf16(false, a.v, false, b,
                                                      (short)0, acc, false, false);
    }
    float bv = bias ? bias[bcol] : 0.f;
    for (int r = 0; r < 8; ++r) {
        int rr = mt * 16 + r + 8 * sel;     // C layout: VGPR r -> M=r (lo half) / M=8+r
        C[rr * N + bcol] = acc[r] + bv;
    }
}

// ---------------------------------------------------------------------------
// Fused pairwise-attention scores:
//   s[i,j] += sum_{b,g} exp( tanh(hid[b,i,:] + sf[b,j,:]) . W_ID[g,:] )
// Grid: 256 blocks = (b, i-tile16, j-tile16); block = 256 threads (8 waves).
// LDS layout: [0,256K)   u-tile   256 x 512 bf16
//             [256K,288K) hid stage 16 x 512 f32  (async DMA'd, VGPR-bypass)
//             [288K,289K) s-accumulator, 256 f32
// ---------------------------------------------------------------------------
#define U_BYTES      (256 * 512 * 2)
#define STAGE_BYTES  (16 * 512 * 4)
#define FUSED_LDS    ((size_t)U_BYTES + STAGE_BYTES + 256 * 4)

__global__ __launch_bounds__(256) void fused_scores(
    const float* __restrict__ hid, const float* __restrict__ sf,
    const unsigned short* __restrict__ WID, float* __restrict__ s)
{
    extern __shared__ char smem[];
    unsigned short* u_us = (unsigned short*)smem;                    // u-tile
    float* hs   = (float*)(smem + U_BYTES);                          // staged hid
    float* sacc = (float*)(smem + U_BYTES + STAGE_BYTES);            // accumulator

    int b  = blockIdx.x >> 4;
    int it = (blockIdx.x >> 2) & 3;
    int jt = blockIdx.x & 3;
    int tid = threadIdx.x;

    sacc[tid] = 0.f;

    const float* hb = hid + (size_t)(b * L_SZ + it * 16) * H_SZ;
    const float* sb = sf  + (size_t)(b * L_SZ + jt * 16) * H_SZ;

    // Stage the 32KB hid tile into LDS with the async DMA engine
    // (GLOBAL_LOAD_ASYNC_TO_LDS_B128, ASYNCcnt-tracked; bypasses VGPRs).
    {
        unsigned stage_off = (unsigned)(uintptr_t)(const void*)hs;   // low 32b = LDS offset
        const char* gsrc = (const char*)hb;
        for (int i = 0; i < STAGE_BYTES / 16 / 256; ++i) {           // 8 x 16B per thread
            int byteoff = (tid + i * 256) * 16;
            unsigned loff = stage_off + (unsigned)byteoff;
            const char* ga = gsrc + byteoff;
            asm volatile("global_load_async_to_lds_b128 %0, %1, off"
                         :: "v"(loff), "v"(ga) : "memory");
        }
        asm volatile("s_wait_asynccnt 0x0" ::: "memory");
    }
    __syncthreads();

    // Phase 1: build u-tile in bf16 (hid from LDS, sf coalesced from L0/L2)
    for (int idx = tid; idx < 256 * 512; idx += 256) {
        int m = idx >> 9;          // 0..255 = i_local*16 + j_local
        int k = idx & 511;
        int il = m >> 4, jl = m & 15;
        float v = tanhf(hs[il * H_SZ + k] + sb[jl * H_SZ + k]);
        u_us[idx] = f2bf(v);
    }
    __syncthreads();

    // Phase 2: WMMA over all (m-tile, g-tile) pairs; exp + reduce over g
    int lane = tid & 31;
    int wave = tid >> 5;
    int sel  = lane >> 4;
    const __bf16* ub = (const __bf16*)u_us;
    const __bf16* Wb = (const __bf16*)WID;
    for (int t = wave; t < 16 * 32; t += 8) {   // uniform trip count -> EXEC all-1
        int mt = t >> 5;          // 16 m-tiles
        int nt = t & 31;          // 32 g-tiles
        int arow = mt * 16 + (lane & 15);
        int bcol = nt * 16 + (lane & 15);
        v8f acc = {};
        for (int kt = 0; kt < 16; ++kt) {
            union { v16bf v; v8bf h[2]; } a;
            const __bf16* pa = ub + arow * H_SZ + kt * 32 + sel * 16;
            a.h[0] = *(const v8bf*)pa;                       // ds_load_b128
            a.h[1] = *(const v8bf*)(pa + 8);
            v16bf bf = *(const v16bf*)(Wb + bcol * H_SZ + kt * 32 + sel * 16);
            acc = __builtin_amdgcn_wmma_f32_16x16x32_bf16(false, a.v, false, bf,
                                                          (short)0, acc, false, false);
        }
        for (int r = 0; r < 8; ++r) {
            float e = __expf(acc[r]);
            e += __shfl_xor(e, 1);
            e += __shfl_xor(e, 2);
            e += __shfl_xor(e, 4);
            e += __shfl_xor(e, 8);       // sum over the 16 g-columns of this tile
            if ((lane & 15) == 0) {
                int m = mt * 16 + r + 8 * sel;
                atomicAdd(&sacc[m], e);
            }
        }
    }
    __syncthreads();

    // Phase 3: one global atomic per (i,j) to accumulate over b
    int m = tid;
    int il = m >> 4, jl = m & 15;
    atomicAdd(&s[(it * 16 + il) * L_SZ + (jt * 16 + jl)], sacc[m]);
}

// ---------------------------------------------------------------------------
// Small elementwise / reduction kernels
// ---------------------------------------------------------------------------
__global__ void f2bf_kernel(const float* __restrict__ src,
                            unsigned short* __restrict__ dst, int n) {
    int i = blockIdx.x * blockDim.x + threadIdx.x;
    if (i < n) dst[i] = f2bf(src[i]);
}

__global__ void zero_kernel(float* __restrict__ p, int n) {
    int i = blockIdx.x * blockDim.x + threadIdx.x;
    if (i < n) p[i] = 0.f;
}

// tsum[b,h] = sum_l tanh(c_slot[b,l,h] + rW[b,h])  (bf16 out for next GEMM)
__global__ void tanhsum_kernel(const float* __restrict__ c_slot,
                               const float* __restrict__ rW,
                               unsigned short* __restrict__ tsum_bf) {
    int idx = blockIdx.x * blockDim.x + threadIdx.x;
    if (idx >= B_SZ * H_SZ) return;
    int b = idx >> 9, h = idx & 511;
    float rv = rW[idx];
    float acc = 0.f;
    for (int l = 0; l < L_SZ; ++l)
        acc += tanhf(c_slot[(size_t)(b * L_SZ + l) * H_SZ + h] + rv);
    tsum_bf[idx] = f2bf(acc);
}

// r_slot = f[b,h] * c_slot[b,l,h]  (fp32 + bf16 copies)
__global__ void rslot_kernel(const float* __restrict__ f,
                             const float* __restrict__ c_slot,
                             float* __restrict__ rslot_f,
                             unsigned short* __restrict__ rslot_bf) {
    int idx = blockIdx.x * blockDim.x + threadIdx.x;
    if (idx >= B_SZ * L_SZ * H_SZ) return;
    float v = f[(idx >> 15) * H_SZ + (idx & 511)] * c_slot[idx];
    rslot_f[idx] = v;
    rslot_bf[idx] = f2bf(v);
}

__global__ void atts_kernel(const float* __restrict__ s, float* __restrict__ atts) {
    int i = threadIdx.x;                // 64 threads
    float sum = 0.f;
    for (int j = 0; j < L_SZ; ++j) sum += s[i * L_SZ + j];
    atts[i] = s[i * L_SZ + i] / sum;
}

// r_inte[b,h] = c_inte[b,h] + sum_i atts[i]*r_slot[b,i,h]
__global__ void rupdate_kernel(const float* __restrict__ atts,
                               const float* __restrict__ rslot_f,
                               const float* __restrict__ c_inte,
                               float* __restrict__ rinte_f,
                               unsigned short* __restrict__ rinte_bf) {
    int b = blockIdx.x;
    for (int h = threadIdx.x; h < H_SZ; h += blockDim.x) {
        float acc = 0.f;
        for (int i = 0; i < L_SZ; ++i)
            acc += atts[i] * rslot_f[(size_t)(b * L_SZ + i) * H_SZ + h];
        float v = acc + c_inte[b * H_SZ + h];
        rinte_f[b * H_SZ + h] = v;
        rinte_bf[b * H_SZ + h] = f2bf(v);
    }
}

__global__ void catinte_kernel(const unsigned short* __restrict__ rinte_bf,
                               const unsigned short* __restrict__ h_bf,
                               unsigned short* __restrict__ out) {
    int idx = blockIdx.x * blockDim.x + threadIdx.x;
    if (idx >= B_SZ * 2 * H_SZ) return;
    int b = idx >> 10, c = idx & 1023;
    out[idx] = (c < H_SZ) ? rinte_bf[b * H_SZ + c]
                          : h_bf[(size_t)(b * L_SZ + (L_SZ - 1)) * H_SZ + (c - H_SZ)];
}

__global__ void catslot_kernel(const unsigned short* __restrict__ h_bf,
                               const unsigned short* __restrict__ rslot_bf,
                               unsigned short* __restrict__ out) {
    int idx = blockIdx.x * blockDim.x + threadIdx.x;
    if (idx >= B_SZ * L_SZ * 2 * H_SZ) return;
    int row = idx >> 10, c = idx & 1023;
    out[idx] = (c < H_SZ) ? h_bf[(size_t)row * H_SZ + c]
                          : rslot_bf[(size_t)row * H_SZ + (c - H_SZ)];
}

// ---------------------------------------------------------------------------
// Host side
// ---------------------------------------------------------------------------
static inline void launch_gemm(const void* A, const void* W, const float* bias,
                               float* C, int M, int N, int K, hipStream_t s) {
    int tiles = (M >> 4) * (N >> 4);
    int blocks = (tiles + 7) / 8;
    wmma_gemm_xt<<<blocks, 256, 0, s>>>((const unsigned short*)A,
                                        (const unsigned short*)W, bias, C, M, N, K);
}

extern "C" void kernel_launch(void* const* d_in, const int* in_sizes, int n_in,
                              void* d_out, int out_size, void* d_ws, size_t ws_size,
                              hipStream_t stream) {
    const float* h      = (const float*)d_in[0];
    const float* c_slot = (const float*)d_in[1];
    const float* c_inte = (const float*)d_in[2];
    const float* W_SF   = (const float*)d_in[3];
    const float* V_SF   = (const float*)d_in[4];
    const float* V1_ID  = (const float*)d_in[5];
    const float* V2_ID_w= (const float*)d_in[6];
    const float* V2_ID_b= (const float*)d_in[7];
    const float* W_ID   = (const float*)d_in[8];
    const float* W_inte = (const float*)d_in[9];
    const float* W_slot = (const float*)d_in[10];

    const int BLH = B_SZ * L_SZ * H_SZ;     // 524288
    const int BH  = B_SZ * H_SZ;            // 8192
    const int HH  = H_SZ * H_SZ;            // 262144
    const int ANS = NLAB * 2 * H_SZ;        // 65536

    char* w = (char*)d_ws;
    size_t off = 0;
    auto carve = [&](size_t bytes) -> void* {
        size_t cur = (off + 255) & ~(size_t)255;
        off = cur + bytes;
        return (void*)(w + cur);
    };

    float* sf_f    = (float*)carve((size_t)BLH * 4);
    float* hid_f   = (float*)carve((size_t)BLH * 4);
    float* rslot_f = (float*)carve((size_t)BLH * 4);
    float* rW_f    = (float*)carve((size_t)BH * 4);
    float* f_f     = (float*)carve((size_t)BH * 4);
    float* rinte_f = (float*)carve((size_t)BH * 4);
    float* s_f     = (float*)carve((size_t)L_SZ * L_SZ * 4);
    float* atts_f  = (float*)carve((size_t)L_SZ * 4);
    unsigned short* h_bf     = (unsigned short*)carve((size_t)BLH * 2);
    unsigned short* rslot_bf = (unsigned short*)carve((size_t)BLH * 2);
    unsigned short* tsum_bf  = (unsigned short*)carve((size_t)BH * 2);
    unsigned short* rinte_bf = (unsigned short*)carve((size_t)BH * 2);
    unsigned short* WSF_b  = (unsigned short*)carve((size_t)HH * 2);
    unsigned short* VSF_b  = (unsigned short*)carve((size_t)HH * 2);
    unsigned short* V1_b   = (unsigned short*)carve((size_t)HH * 2);
    unsigned short* V2_b   = (unsigned short*)carve((size_t)HH * 2);
    unsigned short* WID_b  = (unsigned short*)carve((size_t)HH * 2);
    unsigned short* Wint_b = (unsigned short*)carve((size_t)ANS * 2);
    unsigned short* Wslt_b = (unsigned short*)carve((size_t)ANS * 2);
    unsigned short* cati_b = (unsigned short*)carve((size_t)B_SZ * 2 * H_SZ * 2);
    unsigned short* cats_b = (unsigned short*)carve((size_t)B_SZ * L_SZ * 2 * H_SZ * 2);

    auto conv = [&](const float* src, unsigned short* dst, int n) {
        f2bf_kernel<<<(n + 255) / 256, 256, 0, stream>>>(src, dst, n);
    };

    // One-time bf16 conversions (weights are loop-invariant)
    conv(h, h_bf, BLH);
    conv(W_SF, WSF_b, HH);
    conv(V_SF, VSF_b, HH);
    conv(V1_ID, V1_b, HH);
    conv(V2_ID_w, V2_b, HH);
    conv(W_ID, WID_b, HH);
    conv(W_inte, Wint_b, ANS);
    conv(W_slot, Wslt_b, ANS);
    conv(c_inte, rinte_bf, BH);
    hipMemcpyAsync(rinte_f, c_inte, (size_t)BH * 4, hipMemcpyDeviceToDevice, stream);

    // slot_features = h @ V2_ID_w.T + b   [1024,512]x[512,512]
    launch_gemm(h_bf, V2_b, V2_ID_b, sf_f, B_SZ * L_SZ, H_SZ, H_SZ, stream);

    for (int it = 0; it < ITERS; ++it) {
        // rW = r_inte @ W_SF.T   [16,512]
        launch_gemm(rinte_bf, WSF_b, nullptr, rW_f, B_SZ, H_SZ, H_SZ, stream);
        // tsum = sum_l tanh(c_slot + rW)
        tanhsum_kernel<<<(BH + 255) / 256, 256, 0, stream>>>(c_slot, rW_f, tsum_bf);
        // f = tsum @ V_SF.T      [16,512]
        launch_gemm(tsum_bf, VSF_b, nullptr, f_f, B_SZ, H_SZ, H_SZ, stream);
        // r_slot = f * c_slot
        rslot_kernel<<<(BLH + 255) / 256, 256, 0, stream>>>(f_f, c_slot, rslot_f, rslot_bf);
        // hid = r_slot @ V1_ID.T [1024,512]
        launch_gemm(rslot_bf, V1_b, nullptr, hid_f, B_SZ * L_SZ, H_SZ, H_SZ, stream);
        // s[i,j] = sum_{b,g} exp((tanh(hid_i + sf_j)) . W_ID[g])
        zero_kernel<<<(L_SZ * L_SZ + 255) / 256, 256, 0, stream>>>(s_f, L_SZ * L_SZ);
        fused_scores<<<256, 256, FUSED_LDS, stream>>>(hid_f, sf_f, WID_b, s_f);
        // atts = diag(s)/rowsum(s);  r_inte = c_inte + atts . r_slot
        atts_kernel<<<1, 64, 0, stream>>>(s_f, atts_f);
        rupdate_kernel<<<B_SZ, 256, 0, stream>>>(atts_f, rslot_f, c_inte, rinte_f, rinte_bf);
    }

    // intent_output = [r_inte, h[:,-1]] @ W_inte_ans.T  -> d_out[0:1024]
    catinte_kernel<<<(B_SZ * 2 * H_SZ + 255) / 256, 256, 0, stream>>>(rinte_bf, h_bf, cati_b);
    launch_gemm(cati_b, Wint_b, nullptr, (float*)d_out, B_SZ, NLAB, 2 * H_SZ, stream);

    // slot_output = [h, r_slot] @ W_slot_ans.T -> d_out[1024:66560]
    catslot_kernel<<<(B_SZ * L_SZ * 2 * H_SZ + 255) / 256, 256, 0, stream>>>(h_bf, rslot_bf, cats_b);
    launch_gemm(cats_b, Wslt_b, nullptr, (float*)d_out + B_SZ * NLAB,
                B_SZ * L_SZ, NLAB, 2 * H_SZ, stream);
}
